// Co_occurrence_Infused_Multi_Label_Attention_1606317769004
// MI455X (gfx1250) — compile-verified, matrix operand order checked by reading, not run on hardware
//
#include <hip/hip_runtime.h>
#include <hip/hip_bf16.h>

#define NCLS 8929
#define DM   768
#define TR   512
#define NH   8
#define DK   64
#define NB   4
#define TOK  512
#define CTILES 559          // ceil(8929/16)
#define CPAD  (CTILES*16)   // 8944

typedef __attribute__((ext_vector_type(16))) __bf16 v16bf;
typedef __attribute__((ext_vector_type(8)))  __bf16 v8bf;
typedef __attribute__((ext_vector_type(8)))  float  v8f;
typedef unsigned int u32x4 __attribute__((ext_vector_type(4)));
typedef int          i32x8 __attribute__((ext_vector_type(8)));
typedef int          i32x4 __attribute__((ext_vector_type(4)));

#if __has_builtin(__builtin_amdgcn_tensor_load_to_lds) && __has_builtin(__builtin_amdgcn_s_wait_tensorcnt)
#define USE_TDM 1
#else
#define USE_TDM 0
#endif

__device__ __forceinline__ v8f wmma_bf16(v16bf a, v16bf b, v8f c) {
  // D = A(16x32) * B(32x16) + C, f32 accumulate
  return __builtin_amdgcn_wmma_f32_16x16x32_bf16(false, a, false, b, (short)0, c, false, false);
}

// A-matrix fragment from a plain row-major bf16 matrix (global or LDS).
// ISA 16-bit A layout: lane<16 holds row m=lane, K chunks [kb..kb+8) and [kb+16..kb+24);
// lanes 16-31 hold row m=lane-16, K chunks [kb+8..kb+16) and [kb+24..kb+32).
__device__ __forceinline__ v16bf load_a_frag(const __bf16* p, int ld, int row0, int kbase, int lane) {
  int m = lane & 15, hi = (lane >> 4) & 1;
  const __bf16* r = p + (size_t)(row0 + m) * ld + kbase + hi * 8;
  v8bf lo = *(const v8bf*)r;
  v8bf hv = *(const v8bf*)(r + 16);
  v16bf out;
#pragma unroll
  for (int i = 0; i < 8; ++i) { out[i] = lo[i]; out[8 + i] = hv[i]; }
  return out;
}

// B-matrix fragment from a pre-packed panel: 16 contiguous bf16 per lane (32B).
__device__ __forceinline__ v16bf load_b_frag(const __bf16* tile_base, int lane) {
  return *(const v16bf*)(tile_base + lane * 16);
}

// --------- TDM: DMA one contiguous block (nbytes4 DWORDs) from global to LDS ---------
// D# group0: count=1 | lds_addr | 57-bit global addr | type=2
// D# group1: data_size=4B, 1-D tensor: tensor_dim0 = tile_dim0 = nbytes4, dim1 = 1
__device__ __forceinline__ void tdm_load_block(const void* gsrc, unsigned lds_off, unsigned nbytes4) {
#if USE_TDM
  unsigned long long ga = (unsigned long long)gsrc;
  u32x4 g0;
  g0[0] = 1u;                                               // count=1
  g0[1] = lds_off;                                          // lds_addr (bytes)
  g0[2] = (unsigned)ga;                                     // global_addr[31:0]
  g0[3] = (unsigned)((ga >> 32) & 0x1FFFFFFu) | 0x80000000u; // addr[56:32] | type=2 (bits 127:126)
  i32x8 g1;
  g1[0] = (int)(2u << 16);                                  // data_size=4B; mask/barrier/iterate/pad = 0
  g1[1] = (int)((nbytes4 & 0xFFFFu) << 16);                 // tensor_dim0[15:0]
  g1[2] = (int)((nbytes4 >> 16) | (1u << 16));              // tensor_dim0[31:16] | tensor_dim1 lo = 1
  g1[3] = (int)((nbytes4 & 0xFFFFu) << 16);                 // tensor_dim1 hi=0 | tile_dim0
  g1[4] = 1;                                                // tile_dim1=1, tile_dim2=0
  g1[5] = (int)nbytes4;                                     // tensor_dim0_stride[31:0]
  g1[6] = (int)((nbytes4 & 0xFFFFu) << 16);                 // stride0 hi=0 | dim1_stride lo
  g1[7] = 0;
  i32x4 z4 = {0, 0, 0, 0};
  i32x8 z8 = {0, 0, 0, 0, 0, 0, 0, 0};
  // 6-arg form (clang-23 / therock-10.0 signature)
  __builtin_amdgcn_tensor_load_to_lds(g0, g1, z4, z4, z8, 0);
#else
  (void)gsrc; (void)lds_off; (void)nbytes4;
#endif
}

// Stage a contiguous tile into LDS: TDM path (wave 0 issues, TENSORcnt wait) or
// cooperative-copy fallback. Always ends with a workgroup barrier.
__device__ __forceinline__ void stage_tile(__bf16* lds_dst, const __bf16* gsrc, int n_bf16) {
#if USE_TDM
  if (threadIdx.x < 32) {
    tdm_load_block(gsrc, (unsigned)(size_t)lds_dst, (unsigned)(n_bf16 >> 1));
    __builtin_amdgcn_s_wait_tensorcnt(0);
  }
#else
  for (int i = threadIdx.x; i < (n_bf16 >> 3); i += blockDim.x)
    ((v8bf*)lds_dst)[i] = ((const v8bf*)gsrc)[i];
#endif
  __syncthreads();
}

// ---------------- prep kernels ----------------

__global__ void cvt_kernel(const float* __restrict__ src, __bf16* __restrict__ dst,
                           int n_src, int n_dst) {
  int i = blockIdx.x * blockDim.x + threadIdx.x;
  if (i >= n_dst) return;
  dst[i] = (i < n_src) ? (__bf16)src[i] : (__bf16)0.0f;
}

// Pack W^T into WMMA-B fragment order. B[k][n] = src[n*K + k]. NT = N/16 == 32 here.
__global__ void pack_b_kernel(const float* __restrict__ src, __bf16* __restrict__ dst,
                              int K, int N) {
  int idx = blockIdx.x * blockDim.x + threadIdx.x;
  if (idx >= K * N) return;
  int k = idx % K, n = idx / K;
  float val = src[(size_t)n * K + k];
  int kt = k >> 5, nt = n >> 4, kk = k & 31, nn = n & 15;
  int lane = nn + (((kk >> 3) & 1) << 4);
  int pos  = (kk & 7) + ((kk >> 4) << 3);
  int NT = N >> 4;
  dst[(((size_t)kt * NT + nt) * 32 + lane) * 16 + pos] = (__bf16)val;
}

// ---------------- H-side projections: WK = tanh(H k_w^T + k_b), WV = tanh(H v_w^T + v_b) ----------------
// A-tile (16 x 768) staged to LDS once per block via TDM; 8 waves share it.
// Output pre-swizzled into B-fragment panels:
//   WKfrag[b,z]: B[k=h(64)][n=t(512)]  -> 2 ktiles x 32 ntiles x 512
//   WVfrag[b,z]: B[k=t(512)][n=h(64)]  -> 16 ktiles x 4 ntiles x 512
__global__ void __launch_bounds__(256)
h_proj_kernel(const __bf16* __restrict__ Hbf, const __bf16* __restrict__ kWf,
              const __bf16* __restrict__ vWf, const float* __restrict__ k_b,
              const float* __restrict__ v_b, __bf16* __restrict__ WKfrag,
              __bf16* __restrict__ WVfrag) {
  __shared__ __align__(32) __bf16 hs[16 * DM];   // 24KB A-tile
  int mt = blockIdx.x;                      // 0..127 (2048 token rows / 16)
  int wave = threadIdx.x >> 5, lane = threadIdx.x & 31;
  int row0 = mt * 16;
  int Nl = lane & 15, hi = (lane >> 4) & 1;
  stage_tile(hs, Hbf + (size_t)row0 * DM, 16 * DM);
  for (int which = 0; which < 2; ++which) {
    const __bf16* Wf = which ? vWf : kWf;
    const float* bias = which ? v_b : k_b;
    for (int i = 0; i < 4; ++i) {
      int nt = wave * 4 + i;                // 32 ntiles over TR=512
      v8f acc = {};
      for (int kt = 0; kt < 24; ++kt) {     // K = DM = 768
        v16bf a = load_a_frag(hs, DM, 0, kt * 32, lane);
        v16bf b = load_b_frag(Wf + (size_t)(kt * 32 + nt) * 512, lane);
        acc = wmma_bf16(a, b, acc);
      }
      int col = nt * 16 + Nl;
      float bv = bias[col];
      int z = col >> 6, h = col & 63;
#pragma unroll
      for (int r = 0; r < 8; ++r) {
        int M = r + hi * 8;
        float val = tanhf(acc[r] + bv);
        int trow = row0 + M;
        int bbt = trow >> 9, t = trow & 511;
        __bf16 bvv = (__bf16)val;
        if (which == 0) {
          int kt2 = h >> 5, nt2 = t >> 4, kk = h & 31, nn = t & 15;
          int ln = nn + (((kk >> 3) & 1) << 4);
          int pos = (kk & 7) + ((kk >> 4) << 3);
          WKfrag[(((size_t)bbt * 8 + z) * (2 * 32) + (kt2 * 32 + nt2)) * 512 + ln * 16 + pos] = bvv;
        } else {
          int kt2 = t >> 5, nt2 = h >> 4, kk = t & 31, nn = h & 15;
          int ln = nn + (((kk >> 3) & 1) << 4);
          int pos = (kk & 7) + ((kk >> 4) << 3);
          WVfrag[(((size_t)bbt * 8 + z) * (16 * 4) + (kt2 * 4 + nt2)) * 512 + ln * 16 + pos] = bvv;
        }
      }
    }
  }
}

// ---------------- class-side fused projections ----------------
// Q-tile (16 x 768) staged to LDS via TDM; Qg kept in LDS between stages.
__global__ void __launch_bounds__(256)
class_proj_kernel(const __bf16* __restrict__ Qbf, const __bf16* __restrict__ tWf,
                  const __bf16* __restrict__ qWf, const __bf16* __restrict__ WWf,
                  const float* __restrict__ trans_b, const float* __restrict__ q_b,
                  __bf16* __restrict__ qbf, float* __restrict__ Qw) {
  __shared__ __align__(32) __bf16 qs[16 * DM];   // 24KB A-tile
  __shared__ __align__(32) __bf16 qg[16 * 512];  // 16KB Qg tile
  int ct = blockIdx.x;
  int wave = threadIdx.x >> 5, lane = threadIdx.x & 31;
  int row0 = ct * 16;
  int Nl = lane & 15, hi = (lane >> 4) & 1;
  stage_tile(qs, Qbf + (size_t)row0 * DM, 16 * DM);
  // stage 1: Qg = tanh(Q trans_w^T + trans_b)
  for (int i = 0; i < 4; ++i) {
    int nt = wave * 4 + i;
    v8f acc = {};
    for (int kt = 0; kt < 24; ++kt) {
      v16bf a = load_a_frag(qs, DM, 0, kt * 32, lane);
      v16bf b = load_b_frag(tWf + (size_t)(kt * 32 + nt) * 512, lane);
      acc = wmma_bf16(a, b, acc);
    }
    int col = nt * 16 + Nl;
    float bv = trans_b[col];
#pragma unroll
    for (int r = 0; r < 8; ++r) qg[(r + hi * 8) * 512 + col] = (__bf16)tanhf(acc[r] + bv);
  }
  __syncthreads();
  // stage 2: 64 n-tiles total (32 for q, 32 for W), 8 per wave
  for (int i = 0; i < 8; ++i) {
    int job = wave * 8 + i;
    int which = job >> 5;
    int nt = job & 31;
    const __bf16* Wf = which ? WWf : qWf;
    v8f acc = {};
    for (int kt = 0; kt < 16; ++kt) {
      v16bf a = load_a_frag(qg, 512, 0, kt * 32, lane);
      v16bf b = load_b_frag(Wf + (size_t)(kt * 32 + nt) * 512, lane);
      acc = wmma_bf16(a, b, acc);
    }
    int col = nt * 16 + Nl;
#pragma unroll
    for (int r = 0; r < 8; ++r) {
      int M = r + hi * 8;
      size_t idx = (size_t)(row0 + M) * 512 + col;
      if (which == 0) qbf[idx] = (__bf16)(acc[r] + q_b[col]);
      else            Qw[idx] = acc[r];
    }
  }
}

// ---------------- attention + output ----------------
#define SCLD 520   // padded LDS stride (multiple of 8 elems, rotates banks)
__global__ void __launch_bounds__(32)
attn_kernel(const __bf16* __restrict__ qbf, const float* __restrict__ Qw,
            const __bf16* __restrict__ WKfrag, const __bf16* __restrict__ WVfrag,
            const float* __restrict__ abias, float* __restrict__ out) {
  __shared__ __align__(32) float  sc[16 * SCLD];
  __shared__ __align__(32) __bf16 at[16 * SCLD];
  __shared__ float rmax[2][16], rsum[2][16];
  __shared__ float red[16][16];
  int ct = blockIdx.x, bb = blockIdx.y;
  int lane = threadIdx.x;
  int Nl = lane & 15, hi = (lane >> 4) & 1;
  int row0 = ct * 16;
  float oacc[8];
#pragma unroll
  for (int r = 0; r < 8; ++r) oacc[r] = 0.f;

  for (int z = 0; z < NH; ++z) {
    v16bf aq0 = load_a_frag(qbf, 512, row0, z * 64, lane);
    v16bf aq1 = load_a_frag(qbf, 512, row0, z * 64 + 32, lane);
    const __bf16* WKb = WKfrag + (size_t)(bb * 8 + z) * (2 * 32 * 512);
    // scores[c,t] = q . WK^T  (K = DK = 64)
    for (int nt = 0; nt < 32; ++nt) {
      v8f acc = {};
      acc = wmma_bf16(aq0, load_b_frag(WKb + (size_t)(0 * 32 + nt) * 512, lane), acc);
      acc = wmma_bf16(aq1, load_b_frag(WKb + (size_t)(1 * 32 + nt) * 512, lane), acc);
      int col = nt * 16 + Nl;
      float ab = abias[bb * TOK + col];
#pragma unroll
      for (int r = 0; r < 8; ++r) sc[(r + hi * 8) * SCLD + col] = acc[r] + ab;
    }
    __syncthreads();
    // softmax over t=512 per row; each lane owns (row = Nl, half = hi) of 256 cols
    {
      int rrow = Nl, c0 = hi * 256;
      float m = -1e30f;
      for (int t2 = 0; t2 < 256; ++t2) m = fmaxf(m, sc[rrow * SCLD + c0 + t2]);
      rmax[hi][rrow] = m;
      __syncthreads();
      m = fmaxf(rmax[0][rrow], rmax[1][rrow]);
      float s = 0.f;
      for (int t2 = 0; t2 < 256; ++t2) s += __expf(sc[rrow * SCLD + c0 + t2] - m);
      rsum[hi][rrow] = s;
      __syncthreads();
      float inv = 1.f / (rsum[0][rrow] + rsum[1][rrow]);
      for (int t2 = 0; t2 < 256; ++t2)
        at[rrow * SCLD + c0 + t2] = (__bf16)(__expf(sc[rrow * SCLD + c0 + t2] - m) * inv);
    }
    __syncthreads();
    // context[c,h] = attn . WV  (K = 512), fused with per-row dot against Qw
    const __bf16* WVb = WVfrag + (size_t)(bb * 8 + z) * (16 * 4 * 512);
    for (int nt2 = 0; nt2 < 4; ++nt2) {
      v8f acc = {};
      for (int kt2 = 0; kt2 < 16; ++kt2) {
        if (kt2 < 15) __builtin_prefetch(WVb + (size_t)((kt2 + 1) * 4 + nt2) * 512, 0, 0);
        v16bf aA = load_a_frag(at, SCLD, 0, kt2 * 32, lane);
        v16bf bB = load_b_frag(WVb + (size_t)(kt2 * 4 + nt2) * 512, lane);
        acc = wmma_bf16(aA, bB, acc);
      }
      int h = nt2 * 16 + Nl;
      int qcol = z * 64 + h;
#pragma unroll
      for (int r = 0; r < 8; ++r) {
        int M = r + hi * 8;
        oacc[r] += acc[r] * Qw[(size_t)(row0 + M) * 512 + qcol];
      }
    }
    __syncthreads();
  }
  // reduce oacc across the 16 lanes covering each row
#pragma unroll
  for (int r = 0; r < 8; ++r) red[r + hi * 8][Nl] = oacc[r];
  __syncthreads();
  if (lane < 16) {
    float s = 0.f;
#pragma unroll
    for (int i = 0; i < 16; ++i) s += red[lane][i];
    int c = row0 + lane;
    if (c < NCLS) out[(size_t)bb * NCLS + c] = s;
  }
}

// ---------------- launcher ----------------
extern "C" void kernel_launch(void* const* d_in, const int* in_sizes, int n_in,
                              void* d_out, int out_size, void* d_ws, size_t ws_size,
                              hipStream_t stream) {
  (void)in_sizes; (void)n_in; (void)out_size; (void)ws_size;
  const float* Q       = (const float*)d_in[0];
  const float* H       = (const float*)d_in[1];
  const float* a       = (const float*)d_in[2];
  const float* trans_w = (const float*)d_in[3];
  const float* trans_b = (const float*)d_in[4];
  const float* q_w     = (const float*)d_in[5];
  const float* q_b     = (const float*)d_in[6];
  const float* k_w     = (const float*)d_in[7];
  const float* k_b     = (const float*)d_in[8];
  const float* v_w     = (const float*)d_in[9];
  const float* v_b     = (const float*)d_in[10];
  const float* W_w     = (const float*)d_in[11];
  float* out = (float*)d_out;

  char* p = (char*)d_ws;
  auto alloc = [&](size_t bytes) { char* r = p; p += (bytes + 255) & ~size_t(255); return r; };
  __bf16* Qbf = (__bf16*)alloc((size_t)CPAD * DM * 2);
  __bf16* Hbf = (__bf16*)alloc((size_t)2048 * DM * 2);
  __bf16* tWf = (__bf16*)alloc((size_t)DM * TR * 2);
  __bf16* kWf = (__bf16*)alloc((size_t)DM * TR * 2);
  __bf16* vWf = (__bf16*)alloc((size_t)DM * TR * 2);
  __bf16* qWf = (__bf16*)alloc((size_t)TR * TR * 2);
  __bf16* WWf = (__bf16*)alloc((size_t)TR * TR * 2);
  __bf16* qbf = (__bf16*)alloc((size_t)CPAD * TR * 2);
  float*  Qw  = (float*)alloc((size_t)CPAD * TR * 4);
  __bf16* WKf = (__bf16*)alloc((size_t)NB * NH * 2 * 32 * 512 * 2);
  __bf16* WVf = (__bf16*)alloc((size_t)NB * NH * 16 * 4 * 512 * 2);

  int nQ = CPAD * DM;
  cvt_kernel<<<(nQ + 255) / 256, 256, 0, stream>>>(Q, Qbf, NCLS * DM, nQ);
  int nH = 2048 * DM;
  cvt_kernel<<<(nH + 255) / 256, 256, 0, stream>>>(H, Hbf, nH, nH);
  pack_b_kernel<<<(DM * TR + 255) / 256, 256, 0, stream>>>(trans_w, tWf, DM, TR);
  pack_b_kernel<<<(DM * TR + 255) / 256, 256, 0, stream>>>(k_w, kWf, DM, TR);
  pack_b_kernel<<<(DM * TR + 255) / 256, 256, 0, stream>>>(v_w, vWf, DM, TR);
  pack_b_kernel<<<(TR * TR + 255) / 256, 256, 0, stream>>>(q_w, qWf, TR, TR);
  pack_b_kernel<<<(TR * TR + 255) / 256, 256, 0, stream>>>(W_w, WWf, TR, TR);

  h_proj_kernel<<<128, 256, 0, stream>>>(Hbf, kWf, vWf, k_b, v_b, WKf, WVf);
  class_proj_kernel<<<CTILES, 256, 0, stream>>>(Qbf, tWf, qWf, WWf, trans_b, q_b, qbf, Qw);
  attn_kernel<<<dim3(CTILES, NB), 32, 0, stream>>>(qbf, Qw, WKf, WVf, a, out);
}
